// DlrmTop_91164975825367
// MI455X (gfx1250) — compile-verified, matrix operand order checked by reading, not run on hardware
//
#include <hip/hip_runtime.h>
#include <hip/hip_bf16.h>

typedef __attribute__((ext_vector_type(8)))  float  v8f;
typedef __attribute__((ext_vector_type(16))) __bf16 v16bf;

// ---------------------------------------------------------------------------
// WMMA helpers (CDNA5 gfx1250, wave32)
// ---------------------------------------------------------------------------
__device__ __forceinline__ v8f wmma_bf16(v16bf a, v16bf b, v8f c) {
    // D = A(16x32 bf16) x B(32x16 bf16) + C(16x16 f32)
    return __builtin_amdgcn_wmma_f32_16x16x32_bf16(
        /*neg_a=*/false, a, /*neg_b=*/false, b,
        /*c_mod=*/(short)0, c, /*reuse_a=*/false, /*reuse_b=*/false);
}

// Load a fragment given the per-lane base pointer (row/k-half already applied):
// two contiguous 16-byte loads (K = kb..kb+7 and K = kb+16..kb+23).
__device__ __forceinline__ v16bf load_frag_at(const __bf16* p) {
    uint4 lo = *(const uint4*)(p);
    uint4 hi = *(const uint4*)(p + 16);
    v16bf f;
    __builtin_memcpy(&f, &lo, 16);
    __builtin_memcpy((char*)&f + 16, &hi, 16);
    return f;
}

// Generic fragment load in the ISA 16-bit A/B layout:
//   lane l: row = base_row + (l & 15), k-half = (l>>4)*8
__device__ __forceinline__ v16bf load_frag(const __bf16* base, int ld, int row, int k) {
    const int lane = threadIdx.x & 31;
    return load_frag_at(base + (size_t)(row + (lane & 15)) * ld + k + ((lane >> 4) << 3));
}

// C/D 16x16 f32 layout: lane l -> N = l&15, VGPR v -> M = v + (l<16 ? 0 : 8)
template <typename OutT>
__device__ __forceinline__ void store_c_relu_bias(const v8f& c, OutT* Y,
                                                  int ldy, int mtile, int ntile, float bv) {
    const int lane  = threadIdx.x & 31;
    const int n     = ntile + (lane & 15);
    const int mbase = mtile + ((lane >> 4) << 3);
#pragma unroll
    for (int v = 0; v < 8; ++v) {
        float x = c[v] + bv;
        x = fmaxf(x, 0.0f);
        Y[(size_t)(mbase + v) * ldy + n] = (OutT)x;
    }
}

// ---------------------------------------------------------------------------
// 64 x N x K GEMM:  Y(64xN) = relu( X(64xK,bf16 LDS) * W(NxK,bf16 global)^T + b )
// 8 waves; each wave owns (16*NW)-column chunks (NW N-tiles x 4 M-tiles).
// Force-inlined so X/Y resolve to addrspace(3) (ds_load_b128) and W to
// addrspace(1) (global_load_b128), with K/N/ldx/ldy as literal constants.
// ---------------------------------------------------------------------------
template <typename OutT, int NW>
__device__ __forceinline__ void gemm64(const __bf16* X, int ldx, int K,
                                       const __bf16* W, const float* bias,
                                       int N, OutT* Y, int ldy) {
    const int wid  = threadIdx.x >> 5;
    const int lane = threadIdx.x & 31;
    const int kb   = (lane >> 4) << 3;      // per-lane k-half offset
    const int nchunks = N / (16 * NW);
    for (int nc = wid; nc < nchunks; nc += 8) {
        const int n0 = nc * 16 * NW;
        // hoisted per-lane base pointers (k-loop adds only +k)
        const __bf16* Wl = W + (size_t)(n0 + (lane & 15)) * K + kb;
        const __bf16* Xl = X + (size_t)(lane & 15) * ldx + kb;
        v8f c[4 * NW] = {};
        for (int k = 0; k < K; k += 32) {
            if (k + 128 < K) {  // stream weights ahead of use (L2-resident)
#pragma unroll
                for (int nt = 0; nt < NW; ++nt)
                    __builtin_prefetch(Wl + (size_t)nt * 16 * K + k + 128, 0, 1);
            }
            v16bf b[NW];
#pragma unroll
            for (int nt = 0; nt < NW; ++nt)
                b[nt] = load_frag_at(Wl + (size_t)nt * 16 * K + k);   // global (L2)
#pragma unroll
            for (int mt = 0; mt < 4; ++mt) {
                v16bf a = load_frag_at(Xl + (size_t)mt * 16 * ldx + k);  // LDS
#pragma unroll
                for (int nt = 0; nt < NW; ++nt)
                    c[mt * NW + nt] = wmma_bf16(a, b[nt], c[mt * NW + nt]);
            }
        }
#pragma unroll
        for (int nt = 0; nt < NW; ++nt) {
            const float bv = bias[n0 + nt * 16 + (lane & 15)];
#pragma unroll
            for (int mt = 0; mt < 4; ++mt)
                store_c_relu_bias<OutT>(c[mt * NW + nt], Y, ldy, mt << 4, n0 + nt * 16, bv);
        }
    }
}

// Branchless scatter of gram-matrix lower-triangle entries into the feature
// row: invalid lanes store into the never-read padding slot (index >= 512),
// avoiding per-element exec-mask branches around the divergent ds_store.
__device__ __forceinline__ void extract_tril(const v8f& c, int mt, int nt,
                                             __bf16* xrow) {
    const int lane  = threadIdx.x & 31;
    const int j     = nt + (lane & 15);
    const int ibase = mt + ((lane >> 4) << 3);
#pragma unroll
    for (int v = 0; v < 8; ++v) {
        const int i    = ibase + v;
        const bool ok  = (i < 27) && (j < i);
        const int  idx = ok ? (128 + (i * (i - 1)) / 2 + j) : 516;  // 512..519 = pad
        xrow[idx] = (__bf16)c[v];
    }
}

// ---------------------------------------------------------------------------
// Fused DLRM-top kernel: 64 batch rows per block, 256 threads (8 wave32).
// ---------------------------------------------------------------------------
#define LDX  520    // 512 + 8  (pad -> 1040B row stride, conflict-free b128 loads)
#define LDH  1032   // 1024 + 8
#define LDE  136    // 128 + 8
#define LD4  264    // 256 + 8 (f32)

__shared__ __align__(16) __bf16 sA[64 * LDH];   // x / h2 / h4(f32 view)
__shared__ __align__(16) __bf16 sB[64 * LDH];   // E scratch / h1 / h3

__global__ __launch_bounds__(256, 1)
void dlrm_top_fused(const float* __restrict__ bottom,
                    const __bf16* __restrict__ W0, const __bf16* __restrict__ W1,
                    const __bf16* __restrict__ W2, const __bf16* __restrict__ W3,
                    const float* __restrict__ b0, const float* __restrict__ b1,
                    const float* __restrict__ b2, const float* __restrict__ b3,
                    const float* __restrict__ W4, const float* __restrict__ b4,
                    float* __restrict__ out) {
    const int tid  = threadIdx.x;
    const int wid  = tid >> 5;
    const int lane = tid & 31;
    const int row0 = blockIdx.x * 64;

    // ---- Stage 1: features (bottom copy + WMMA gram matrix tril) ----
    for (int rr = 0; rr < 8; ++rr) {
        const int r = wid * 8 + rr;                        // local row
        const float* src = bottom + (size_t)(row0 + r) * 27 * 128;
        __bf16* E = sB + wid * (32 * LDE);                 // per-wave 32x128 bf16

        // load + convert 27x128 f32 -> bf16 (864 float4 per row)
        for (int i = lane; i < 864; i += 32) {
            float4 f = ((const float4*)src)[i];
            const int rI = (i * 4) >> 7, cI = (i * 4) & 127;
            __bf16* e = E + rI * LDE + cI;
            e[0] = (__bf16)f.x; e[1] = (__bf16)f.y; e[2] = (__bf16)f.z; e[3] = (__bf16)f.w;
        }
        // zero pad rows 27..31
        for (int i = 27 * 128 + lane * 4; i < 32 * 128; i += 128) {
            __bf16* e = E + (i >> 7) * LDE + (i & 127);
            e[0] = (__bf16)0.f; e[1] = (__bf16)0.f; e[2] = (__bf16)0.f; e[3] = (__bf16)0.f;
        }
        // bottom_mlp_output copy (feature 0..127)
        {
            float4 f = ((const float4*)src)[lane];
            __bf16* xr = sA + (size_t)r * LDX + lane * 4;
            xr[0] = (__bf16)f.x; xr[1] = (__bf16)f.y; xr[2] = (__bf16)f.z; xr[3] = (__bf16)f.w;
        }
        // zero feature tail (479 zero-pad + K padding to 512)
        for (int idx = 479 + lane; idx < 512; idx += 32)
            sA[(size_t)r * LDX + idx] = (__bf16)0.f;

        // gram: G = E * E^T ; A-frag and B-frag layouts coincide -> reuse frags
        v8f c00 = {}, c10 = {}, c11 = {};
        for (int k = 0; k < 128; k += 32) {
            v16bf f0 = load_frag(E, LDE, 0, k);
            v16bf f1 = load_frag(E, LDE, 16, k);
            c00 = wmma_bf16(f0, f0, c00);
            c10 = wmma_bf16(f1, f0, c10);
            c11 = wmma_bf16(f1, f1, c11);
        }
        __bf16* xrow = sA + (size_t)r * LDX;
        extract_tril(c00, 0, 0, xrow);
        extract_tril(c10, 16, 0, xrow);
        extract_tril(c11, 16, 16, xrow);
    }
    __syncthreads();

    // ---- Stage 2: MLP (bf16 WMMA, f32 accumulate, fused bias+ReLU) ----
    gemm64<__bf16, 4>(sA, LDX, 512,  W0, b0, 1024, sB, LDH);   // x  -> h1
    __syncthreads();
    gemm64<__bf16, 4>(sB, LDH, 1024, W1, b1, 1024, sA, LDH);   // h1 -> h2
    __syncthreads();
    gemm64<__bf16, 4>(sA, LDH, 1024, W2, b2, 512,  sB, LDX);   // h2 -> h3
    __syncthreads();
    float* H4 = (float*)sA;                                     // h4 kept in f32
    gemm64<float, 2>(sB, LDX, 512,  W3, b3, 256,  H4, LD4);    // h3 -> h4
    __syncthreads();

    // ---- Stage 3: final 256 -> 1 dot product (f32) ----
    {
        const int r = tid >> 2, seg = tid & 3;
        const float* h = H4 + (size_t)r * LD4 + seg * 64;
        const float* w = W4 + seg * 64;
        float s = 0.f;
#pragma unroll 16
        for (int k = 0; k < 64; ++k) s = fmaf(h[k], w[k], s);
        s += __shfl_xor(s, 1, 32);
        s += __shfl_xor(s, 2, 32);
        if (seg == 0) out[row0 + r] = s + b4[0];
    }
}

// ---------------------------------------------------------------------------
// Weight f32 -> bf16 conversion with K zero-padding
// ---------------------------------------------------------------------------
__global__ __launch_bounds__(256)
void cvt_pad_kernel(const float* __restrict__ src, __bf16* __restrict__ dst,
                    int rows, int kin, int kout) {
    const int idx = blockIdx.x * blockDim.x + threadIdx.x;
    if (idx >= rows * kout) return;
    const int r = idx / kout, k = idx - r * kout;
    dst[idx] = (k < kin) ? (__bf16)src[(size_t)r * kin + k] : (__bf16)0.0f;
}

// ---------------------------------------------------------------------------
extern "C" void kernel_launch(void* const* d_in, const int* in_sizes, int n_in,
                              void* d_out, int out_size, void* d_ws, size_t ws_size,
                              hipStream_t stream) {
    const float* bottom = (const float*)d_in[0];
    const float* W0 = (const float*)d_in[1];
    const float* b0 = (const float*)d_in[2];
    const float* W1 = (const float*)d_in[3];
    const float* b1 = (const float*)d_in[4];
    const float* W2 = (const float*)d_in[5];
    const float* b2 = (const float*)d_in[6];
    const float* W3 = (const float*)d_in[7];
    const float* b3 = (const float*)d_in[8];
    const float* W4 = (const float*)d_in[9];
    const float* b4 = (const float*)d_in[10];
    float* out = (float*)d_out;

    // workspace: bf16 weights (W0 K-padded 480->512); ~4.25 MB total
    __bf16* W0bf = (__bf16*)d_ws;             // 1024 x 512
    __bf16* W1bf = W0bf + 1024 * 512;         // 1024 x 1024
    __bf16* W2bf = W1bf + 1024 * 1024;        // 512 x 1024
    __bf16* W3bf = W2bf + 512 * 1024;         // 256 x 512

    auto cvt = [&](const float* s, __bf16* d, int rows, int kin, int kout) {
        const int total = rows * kout;
        cvt_pad_kernel<<<(total + 255) / 256, 256, 0, stream>>>(s, d, rows, kin, kout);
    };
    cvt(W0, W0bf, 1024, 480, 512);
    cvt(W1, W1bf, 1024, 1024, 1024);
    cvt(W2, W2bf, 512, 1024, 1024);
    cvt(W3, W3bf, 256, 512, 512);

    dlrm_top_fused<<<32768 / 64, 256, 0, stream>>>(
        bottom, W0bf, W1bf, W2bf, W3bf, b0, b1, b2, b3, W4, b4, out);
}
